// MoELMHead_82892868813389
// MI455X (gfx1250) — matrix operand order repaired and checked by reading, not compiled, not run on hardware
//
#include <hip/hip_runtime.h>

#define VOCAB   32000
#define HIDDEN  2048
#define NEXP    16
#define EV      2000   // vocab slice per expert
#define NTOK    4096

typedef __attribute__((ext_vector_type(16))) __bf16        v16bf;
typedef __attribute__((ext_vector_type(8)))  float         v8f;
typedef __attribute__((ext_vector_type(4)))  unsigned int  u32x4;

union BF16x16 { v16bf v; u32x4 q[2]; unsigned int w[8]; };

__device__ __forceinline__ unsigned short f2bf_rne(float f) {
    unsigned int b = __float_as_uint(f);
    b += 0x7FFFu + ((b >> 16) & 1u);
    return (unsigned short)(b >> 16);
}

// Truncating pack: dword = { bf16(hi), bf16(lo) } in ONE v_perm_b32.
__device__ __forceinline__ unsigned int pkbf_trunc(float hi, float lo) {
    return __builtin_amdgcn_perm(__float_as_uint(hi), __float_as_uint(lo), 0x07060302u);
}

// ---------------------------------------------------------------- init
__global__ void moe_init_kernel(int* __restrict__ counts) {
    if (threadIdx.x < NEXP) counts[threadIdx.x] = 0;
}

// ---------------------------------------------------------------- fp32 -> bf16 pre-convert
// 8 elements / thread: 2x b128 load, RNE pack, 1x b128 store.
__global__ __launch_bounds__(256)
void cvt_bf16_kernel(const float* __restrict__ src, uint4* __restrict__ dst, long n8) {
    long i = (long)blockIdx.x * blockDim.x + threadIdx.x;
    if (i >= n8) return;
    const float4* s = (const float4*)(src + i * 8);
    float4 x = s[0], y = s[1];
    uint4 o;
    o.x = ((unsigned)f2bf_rne(x.y) << 16) | f2bf_rne(x.x);
    o.y = ((unsigned)f2bf_rne(x.w) << 16) | f2bf_rne(x.z);
    o.z = ((unsigned)f2bf_rne(y.y) << 16) | f2bf_rne(y.x);
    o.w = ((unsigned)f2bf_rne(y.w) << 16) | f2bf_rne(y.z);
    dst[i] = o;
}

// ---------------------------------------------------------------- router
__global__ __launch_bounds__(256)
void moe_router_kernel(const float* __restrict__ hs,
                       const float* __restrict__ rw,
                       int* __restrict__ counts,
                       int* __restrict__ lists,
                       unsigned int* __restrict__ sel) {
    const int t   = blockIdx.x;
    const int tid = threadIdx.x;
    const int e   = tid >> 4;
    const int sub = tid & 15;
    __shared__ float part[256];
    __shared__ float logits[NEXP];

    const float* h = hs + (size_t)t * HIDDEN;
    const float* w = rw + (size_t)e * HIDDEN;
    float s = 0.f;
    for (int i = sub; i < HIDDEN; i += 16) s += h[i] * w[i];
    part[tid] = s;
    __syncthreads();
    if (sub == 0) {
        float acc = 0.f;
        #pragma unroll
        for (int i = 0; i < 16; ++i) acc += part[(e << 4) + i];
        logits[e] = acc;
    }
    __syncthreads();
    if (tid == 0) {
        int i1 = 0; float m1 = logits[0];
        #pragma unroll
        for (int i = 1; i < NEXP; ++i)
            if (logits[i] > m1) { m1 = logits[i]; i1 = i; }
        int i2 = 0; float m2 = -3.402823466e38f;
        #pragma unroll
        for (int i = 0; i < NEXP; ++i)
            if (i != i1 && logits[i] > m2) { m2 = logits[i]; i2 = i; }
        sel[t] = (1u << i1) | (1u << i2);
        int p1 = atomicAdd(&counts[i1], 1); lists[i1 * NTOK + p1] = t;
        int p2 = atomicAdd(&counts[i2], 1); lists[i2 * NTOK + p2] = t;
    }
}

// ---------------------------------------------------------------- -inf fill
__global__ __launch_bounds__(256)
void moe_fill_kernel(const unsigned int* __restrict__ sel,
                     float* __restrict__ out) {
    const int t = blockIdx.x;
    const int e = blockIdx.y;
    if (sel[t] & (1u << e)) return;
    const float ninf = -__builtin_inff();
    float4* p = (float4*)(out + (size_t)t * VOCAB + (size_t)e * EV);
    const float4 v = make_float4(ninf, ninf, ninf, ninf);
    for (int i = threadIdx.x; i < EV / 4; i += blockDim.x) p[i] = v;
}

// ---------------------------------------------------------------- shared GEMM epilogue
__device__ __forceinline__ void gemm_store(const int* __restrict__ list, int cnt, int e,
                                           int mbase, int nbase, int lhalf, int l16,
                                           const v8f& acc00, const v8f& acc01,
                                           const v8f& acc10, const v8f& acc11,
                                           float* __restrict__ out) {
    const int ncol0 = nbase + l16;
    const int ncol1 = nbase + 16 + l16;
    #pragma unroll
    for (int v = 0; v < 8; ++v) {
        const int ml = v + lhalf * 8;
        const int ra = mbase + ml;
        const int rb = mbase + 16 + ml;
        if (ra < cnt) {
            const size_t base = (size_t)list[ra] * VOCAB + (size_t)e * EV;
            if (ncol0 < EV) out[base + ncol0] = acc00[v];
            if (ncol1 < EV) out[base + ncol1] = acc01[v];
        }
        if (rb < cnt) {
            const size_t base = (size_t)list[rb] * VOCAB + (size_t)e * EV;
            if (ncol0 < EV) out[base + ncol0] = acc10[v];
            if (ncol1 < EV) out[base + ncol1] = acc11[v];
        }
    }
}

// ---------------------------------------------------------------- grouped GEMM, bf16 operands
// grid = (16, 64, 16); 8 waves/block as 2(M) x 4(N); wave = 32M x 32N.
// Inner loop: 8x global_load_b128 + 4x v_wmma_f32_16x16x32_bf16, zero convert VALU.
__global__ __launch_bounds__(256)
void moe_gemm_bf16_kernel(const unsigned short* __restrict__ hsb,
                          const unsigned short* __restrict__ ewb,
                          const int* __restrict__ counts,
                          const int* __restrict__ lists,
                          float* __restrict__ out) {
    const int e   = blockIdx.z;
    const int cnt = counts[e];
    const int m0  = blockIdx.y * 64;
    if (m0 >= cnt) return;
    const int n0    = blockIdx.x * 128;
    const int wave  = threadIdx.x >> 5;
    const int lane  = threadIdx.x & 31;
    const int mbase = m0 + (wave & 1) * 32;
    const int nbase = n0 + (wave >> 1) * 32;
    const int lhalf = lane >> 4;
    const int l16   = lane & 15;

    const int* list = lists + e * NTOK;

    const int rc0 = min(mbase + l16,      cnt - 1);
    const int rc1 = min(mbase + 16 + l16, cnt - 1);
    const unsigned short* a0p = hsb + (size_t)list[rc0] * HIDDEN;
    const unsigned short* a1p = hsb + (size_t)list[rc1] * HIDDEN;

    const int nc0 = min(nbase + l16,      EV - 1);
    const int nc1 = min(nbase + 16 + l16, EV - 1);
    const unsigned short* b0p = ewb + ((size_t)e * EV + nc0) * HIDDEN;
    const unsigned short* b1p = ewb + ((size_t)e * EV + nc1) * HIDDEN;

    v8f acc00 = {}, acc01 = {}, acc10 = {}, acc11 = {};

    #pragma unroll 2
    for (int k = 0; k < HIDDEN; k += 32) {
        const int ca = k + lhalf * 16;   // A frag: halves [ca..ca+15]
        const int cb = k + lhalf * 16;   // B frag: halves [cb..cb+15]

        __builtin_prefetch(b0p + cb + 64, 0, 0);
        __builtin_prefetch(b1p + cb + 64, 0, 0);

        BF16x16 a0, a1, b0, b1;
        a0.q[0] = *(const u32x4*)(a0p + ca);
        a0.q[1] = *(const u32x4*)(a0p + ca + 8);
        a1.q[0] = *(const u32x4*)(a1p + ca);
        a1.q[1] = *(const u32x4*)(a1p + ca + 8);
        b0.q[0] = *(const u32x4*)(b0p + cb);
        b0.q[1] = *(const u32x4*)(b0p + cb + 8);
        b1.q[0] = *(const u32x4*)(b1p + cb);
        b1.q[1] = *(const u32x4*)(b1p + cb + 8);

        acc00 = __builtin_amdgcn_wmma_f32_16x16x32_bf16(false, a0.v, false, b0.v,
                                                        (short)0, acc00, false, false);
        acc01 = __builtin_amdgcn_wmma_f32_16x16x32_bf16(false, a0.v, false, b1.v,
                                                        (short)0, acc01, false, false);
        acc10 = __builtin_amdgcn_wmma_f32_16x16x32_bf16(false, a1.v, false, b0.v,
                                                        (short)0, acc10, false, false);
        acc11 = __builtin_amdgcn_wmma_f32_16x16x32_bf16(false, a1.v, false, b1.v,
                                                        (short)0, acc11, false, false);
    }
    gemm_store(list, cnt, e, mbase, nbase, lhalf, l16, acc00, acc01, acc10, acc11, out);
}

// ---------------------------------------------------------------- fallback GEMM, fp32 operands
// Used only if workspace can't hold bf16 copies. In-loop convert via v_perm_b32
// truncation: 1 VALU per 2 elements.
__device__ __forceinline__ void cv8t(const float* __restrict__ p, unsigned int* d) {
    float4 x = ((const float4*)p)[0];
    float4 y = ((const float4*)p)[1];
    d[0] = pkbf_trunc(x.y, x.x);
    d[1] = pkbf_trunc(x.w, x.z);
    d[2] = pkbf_trunc(y.y, y.x);
    d[3] = pkbf_trunc(y.w, y.z);
}

__global__ __launch_bounds__(256)
void moe_gemm_f32_kernel(const float* __restrict__ hs,
                         const float* __restrict__ ew,
                         const int* __restrict__ counts,
                         const int* __restrict__ lists,
                         float* __restrict__ out) {
    const int e   = blockIdx.z;
    const int cnt = counts[e];
    const int m0  = blockIdx.y * 64;
    if (m0 >= cnt) return;
    const int n0    = blockIdx.x * 128;
    const int wave  = threadIdx.x >> 5;
    const int lane  = threadIdx.x & 31;
    const int mbase = m0 + (wave & 1) * 32;
    const int nbase = n0 + (wave >> 1) * 32;
    const int lhalf = lane >> 4;
    const int l16   = lane & 15;

    const int* list = lists + e * NTOK;

    const int rc0 = min(mbase + l16,      cnt - 1);
    const int rc1 = min(mbase + 16 + l16, cnt - 1);
    const float* a0p = hs + (size_t)list[rc0] * HIDDEN;
    const float* a1p = hs + (size_t)list[rc1] * HIDDEN;

    const int nc0 = min(nbase + l16,      EV - 1);
    const int nc1 = min(nbase + 16 + l16, EV - 1);
    const float* b0p = ew + ((size_t)e * EV + nc0) * HIDDEN;
    const float* b1p = ew + ((size_t)e * EV + nc1) * HIDDEN;

    v8f acc00 = {}, acc01 = {}, acc10 = {}, acc11 = {};

    for (int k = 0; k < HIDDEN; k += 32) {
        const int ca = k + lhalf * 16;
        const int cb = k + lhalf * 16;

        __builtin_prefetch(b0p + cb + 32, 0, 0);
        __builtin_prefetch(b1p + cb + 32, 0, 0);

        BF16x16 a0, a1, b0, b1;
        cv8t(a0p + ca,      a0.w);  cv8t(a0p + ca + 8,  a0.w + 4);
        cv8t(a1p + ca,      a1.w);  cv8t(a1p + ca + 8,  a1.w + 4);
        cv8t(b0p + cb,      b0.w);  cv8t(b0p + cb + 8,  b0.w + 4);
        cv8t(b1p + cb,      b1.w);  cv8t(b1p + cb + 8,  b1.w + 4);

        acc00 = __builtin_amdgcn_wmma_f32_16x16x32_bf16(false, a0.v, false, b0.v,
                                                        (short)0, acc00, false, false);
        acc01 = __builtin_amdgcn_wmma_f32_16x16x32_bf16(false, a0.v, false, b1.v,
                                                        (short)0, acc01, false, false);
        acc10 = __builtin_amdgcn_wmma_f32_16x16x32_bf16(false, a1.v, false, b0.v,
                                                        (short)0, acc10, false, false);
        acc11 = __builtin_amdgcn_wmma_f32_16x16x32_bf16(false, a1.v, false, b1.v,
                                                        (short)0, acc11, false, false);
    }
    gemm_store(list, cnt, e, mbase, nbase, lhalf, l16, acc00, acc01, acc10, acc11, out);
}

// ---------------------------------------------------------------- launch
extern "C" void kernel_launch(void* const* d_in, const int* in_sizes, int n_in,
                              void* d_out, int out_size, void* d_ws, size_t ws_size,
                              hipStream_t stream) {
    const float* hs = (const float*)d_in[0];   // [4096, 2048]
    const float* ew = (const float*)d_in[1];   // [16, 2000, 2048]
    const float* rw = (const float*)d_in[2];   // [16, 2048]
    float* out = (float*)d_out;                // [4096, 32000]

    char* ws = (char*)d_ws;
    int*          counts = (int*)ws;                       // 16 ints
    unsigned int* sel    = (unsigned int*)(ws + 64);       // 4096 uints
    int*          lists  = (int*)(ws + 64 + NTOK * 4);     // 16*4096 ints

    const size_t META   = 64 + (size_t)NTOK * 4 + (size_t)NEXP * NTOK * 4;  // 278592, 16B-aligned
    const size_t HSB_SZ = (size_t)NTOK * HIDDEN * 2;                        // 16 MiB
    const size_t EWB_SZ = (size_t)NEXP * EV * HIDDEN * 2;                   // 125 MiB
    const bool   big_ws = ws_size >= META + HSB_SZ + EWB_SZ;

    moe_init_kernel<<<1, 32, 0, stream>>>(counts);
    moe_router_kernel<<<NTOK, 256, 0, stream>>>(hs, rw, counts, lists, sel);
    moe_fill_kernel<<<dim3(NTOK, NEXP), 256, 0, stream>>>(sel, out);

    if (big_ws) {
        unsigned short* hsb = (unsigned short*)(ws + META);
        unsigned short* ewb = (unsigned short*)(ws + META + HSB_SZ);
        const long n8_h = (long)NTOK * HIDDEN / 8;
        const long n8_w = (long)NEXP * EV * HIDDEN / 8;
        cvt_bf16_kernel<<<(int)((n8_h + 255) / 256), 256, 0, stream>>>(hs, (uint4*)hsb, n8_h);
        cvt_bf16_kernel<<<(int)((n8_w + 255) / 256), 256, 0, stream>>>(ew, (uint4*)ewb, n8_w);
        moe_gemm_bf16_kernel<<<dim3(16, NTOK / 64, NEXP), 256, 0, stream>>>(
            hsb, ewb, counts, lists, out);
    } else {
        moe_gemm_f32_kernel<<<dim3(16, NTOK / 64, NEXP), 256, 0, stream>>>(
            hs, ew, counts, lists, out);
    }
}
